// CNN_ODE_1340029796568
// MI455X (gfx1250) — matrix operand order repaired and checked by reading, not compiled, not run on hardware
//
#include <hip/hip_runtime.h>

#define BATCH  65536
#define SEQL   40
#define INDIM  24
#define NKER   36
#define KFLAT  1440   /* NKER*SEQL */
#define ENCD   128
#define HID    64
#define STEPS  50

typedef __attribute__((ext_vector_type(16))) _Float16 v16h;
typedef __attribute__((ext_vector_type(2)))  __fp16   v2fp;
typedef __attribute__((ext_vector_type(8)))  float    v8f;
typedef __attribute__((ext_vector_type(4)))  unsigned v4u;
typedef __attribute__((ext_vector_type(8)))  unsigned v8u;
typedef __attribute__((ext_vector_type(2)))  unsigned v2u;

__device__ __forceinline__ v8f wmma_f16(v16h a, v16h b, v8f c){
  return __builtin_amdgcn_wmma_f32_16x16x32_f16(false, a, false, b, (short)0, c, false, false);
}

__device__ __forceinline__ unsigned pk2(float a, float b){
  v2fp h = __builtin_amdgcn_cvt_pkrtz(a, b);
  return __builtin_bit_cast(unsigned, h);
}

__device__ __forceinline__ v16h combine16(v4u a, v4u b){
  v8u u; u[0]=a[0]; u[1]=a[1]; u[2]=a[2]; u[3]=a[3];
         u[4]=b[0]; u[5]=b[1]; u[6]=b[2]; u[7]=b[3];
  return __builtin_bit_cast(v16h, u);
}

__device__ __forceinline__ float ftanh(float x){
#if __has_builtin(__builtin_amdgcn_tanhf)
  return __builtin_amdgcn_tanhf(x);            // v_tanh_f32 (gfx1250)
#else
  // tanh(x) = 1 - 2/(exp(2x)+1), via v_exp_f32 (base-2) + v_rcp_f32
  float t = __builtin_amdgcn_exp2f(x * 2.8853900817779268f);
  return 1.0f - 2.0f * __builtin_amdgcn_rcpf(t + 1.0f);
#endif
}
__device__ __forceinline__ float fsilu(float x){
  float t = __builtin_amdgcn_exp2f(x * -1.4426950408889634f);
  return x * __builtin_amdgcn_rcpf(1.0f + t);
}

// A-operand fragment (16x32 f16) from row-major f16 weight [M x ldk].
// lane lo: halves j -> K=32kc+j (j<16); hi shifted by 16.
__device__ __forceinline__ v16h load_afrag(const _Float16* W, int ldk, int mt, int kc,
                                           int lm, int hi){
  const _Float16* p = W + (size_t)(mt*16 + lm)*ldk + kc*32 + hi*16;
  v4u a = *(const v4u*)p;
  v4u b = *(const v4u*)(p + 8);
  return combine16(a, b);
}

// Build B-operand fragment (32x16 f16) from two f32 D-layout tiles
// (t0 = K rows 0..15, t1 = K rows 16..31). Only a lane<->lane+16 half swap needed:
// v_permlane16_swap_b32 does it in one instruction per register pair.
__device__ __forceinline__ v16h make_bfrag(v8f t0, v8f t1, int hi){
  unsigned own0[4], own1[4];
  #pragma unroll
  for(int v=0; v<4; ++v){
    own0[v] = pk2(t0[2*v], t0[2*v+1]);
    own1[v] = pk2(t1[2*v], t1[2*v+1]);
  }
  v8u u;
#if __has_builtin(__builtin_amdgcn_permlane16_swap)
  (void)hi;
  #pragma unroll
  for(int v=0; v<4; ++v){
    // new_a[16:31] = old_b[0:15]; new_b[0:15] = old_a[16:31]
    v2u r = __builtin_amdgcn_permlane16_swap(own0[v], own1[v], false, false);
    u[v]   = r[0];
    u[v+4] = r[1];
  }
#else
  unsigned oth0[4], oth1[4];
  #pragma unroll
  for(int v=0; v<4; ++v){
    oth0[v] = (unsigned)__shfl_xor((int)own0[v], 16, 32);
    oth1[v] = (unsigned)__shfl_xor((int)own1[v], 16, 32);
  }
  #pragma unroll
  for(int v=0; v<4; ++v){
    u[v]   = hi ? oth1[v] : own0[v];
    u[v+4] = hi ? own1[v] : oth0[v];
  }
#endif
  return __builtin_bit_cast(v16h, u);
}

__device__ __forceinline__ v8f bias_tile(const float* b, int mt, int hi){
  v8f r; int s0 = mt*16 + hi*8;
  #pragma unroll
  for(int i=0;i<8;++i) r[i] = b[s0+i];
  return r;
}
__device__ __forceinline__ v8f bias_tile_c(const float* b, int mt, int hi, int nmax){
  v8f r; int s0 = mt*16 + hi*8;
  #pragma unroll
  for(int i=0;i<8;++i){ int s = s0+i; r[i] = (s < nmax) ? b[s] : 0.0f; }
  return r;
}

// Conv im2col B-fragment: 16 halves at K = 8*kb8 .. +15 of the patch
// patch[K] = x[b, s-1+K/24, K%24]; 8-half segments never cross a row (24 = 3*8).
__device__ __forceinline__ v16h conv_bfrag(const float* xb, int s, int kb8){
  float v[16];
  #pragma unroll
  for(int sg=0; sg<2; ++sg){
    int g   = kb8 + sg;
    int row = (g*11) >> 5;          // g/3 for g<12
    int c0  = (g - row*3) * 8;
    int sp  = s - 1 + row;
    if (g < 9 && sp >= 0 && sp < SEQL){
      const float4* p = (const float4*)(xb + sp*INDIM + c0);
      float4 A = p[0]; float4 Bv = p[1];
      v[sg*8+0]=A.x;  v[sg*8+1]=A.y;  v[sg*8+2]=A.z;  v[sg*8+3]=A.w;
      v[sg*8+4]=Bv.x; v[sg*8+5]=Bv.y; v[sg*8+6]=Bv.z; v[sg*8+7]=Bv.w;
    } else {
      #pragma unroll
      for(int t=0;t<8;++t) v[sg*8+t] = 0.0f;
    }
  }
  v8u u;
  #pragma unroll
  for(int t=0;t<8;++t) u[t] = pk2(v[2*t], v[2*t+1]);
  return __builtin_bit_cast(v16h, u);
}

// ---------------- prep kernels ----------------
__global__ void cvt_f16_kernel(const float* __restrict__ s, _Float16* __restrict__ d, int n){
  int i = blockIdx.x*blockDim.x + threadIdx.x;
  if (i < n) d[i] = (_Float16)s[i];
}
// conv_w [36][24][3] -> padded [48][96] f16, K = kpos*24 + c
__global__ void prep_convw_kernel(const float* __restrict__ cw, _Float16* __restrict__ wA){
  int idx = blockIdx.x*blockDim.x + threadIdx.x;
  if (idx >= 48*96) return;
  int nk = idx / 96, K = idx - nk*96;
  float v = 0.0f;
  if (nk < NKER && K < 72){
    int kpos = K / 24, c = K - kpos*24;
    v = cw[(nk*INDIM + c)*3 + kpos];
  }
  wA[idx] = (_Float16)v;
}

// ---------------- frontend: conv + silu + enc1 + relu + enc2 ----------------
__global__ __launch_bounds__(32, 1) void frontend_kernel(
    const float* __restrict__ x, const float* __restrict__ convb,
    const _Float16* __restrict__ wA, const _Float16* __restrict__ e1h,
    const float* __restrict__ e1b, const _Float16* __restrict__ e2h,
    const float* __restrict__ e2b, float* __restrict__ theta0T)
{
  __shared__ __align__(16) _Float16 act[16*KFLAT];
  const int lane = threadIdx.x & 31;
  const int n  = lane & 15;
  const int hi = lane >> 4;
  const int lm = n;
  const int b0 = blockIdx.x * 16;
  const float* xb = x + (size_t)(b0 + n) * (SEQL*INDIM);

  // conv weight A-frags (constant over s)
  v16h ac[3][3];
  #pragma unroll
  for(int mt=0;mt<3;++mt)
    #pragma unroll
    for(int kc=0;kc<3;++kc) ac[mt][kc] = load_afrag(wA, 96, mt, kc, lm, hi);
  v8f cbt[3];
  #pragma unroll
  for(int mt=0;mt<3;++mt) cbt[mt] = bias_tile_c(convb, mt, hi, NKER);

  #pragma unroll 1
  for(int s=0; s<SEQL; ++s){
    v16h bf0 = conv_bfrag(xb, s, 0 + hi*2);
    v16h bf1 = conv_bfrag(xb, s, 4 + hi*2);
    v16h bf2 = conv_bfrag(xb, s, 8 + hi*2);
    #pragma unroll
    for(int mt=0;mt<3;++mt){
      v8f c = cbt[mt];
      c = wmma_f16(ac[mt][0], bf0, c);
      c = wmma_f16(ac[mt][1], bf1, c);
      c = wmma_f16(ac[mt][2], bf2, c);
      #pragma unroll
      for(int i=0;i<8;++i){
        int nk = mt*16 + hi*8 + i;
        if (nk < NKER) act[n*KFLAT + nk*SEQL + s] = (_Float16)fsilu(c[i]);
      }
    }
  }
  __syncthreads();

  // enc1: [128 x 1440] @ actT, 45 K-chunks x 8 M-tiles
  v8f e1[8];
  #pragma unroll
  for(int mt=0;mt<8;++mt) e1[mt] = bias_tile(e1b, mt, hi);
  #pragma unroll 1
  for(int kc=0; kc<45; ++kc){
    const v4u* pl = (const v4u*)&act[n*KFLAT + kc*32 + hi*16];
    v16h bf = combine16(pl[0], pl[1]);
    #pragma unroll
    for(int mt=0;mt<8;++mt){
      v16h af = load_afrag(e1h, KFLAT, mt, kc, lm, hi);
      e1[mt] = wmma_f16(af, bf, e1[mt]);
    }
  }
  #pragma unroll
  for(int mt=0;mt<8;++mt)
    #pragma unroll
    for(int i=0;i<8;++i) e1[mt][i] = fmaxf(e1[mt][i], 0.0f);

  // enc2: [64 x 128]
  v16h bt[4];
  #pragma unroll
  for(int c=0;c<4;++c) bt[c] = make_bfrag(e1[2*c], e1[2*c+1], hi);
  #pragma unroll
  for(int mt=0;mt<4;++mt){
    v8f cc = bias_tile(e2b, mt, hi);
    #pragma unroll
    for(int c=0;c<4;++c) cc = wmma_f16(load_afrag(e2h, ENCD, mt, c, lm, hi), bt[c], cc);
    #pragma unroll
    for(int i=0;i<8;++i)
      theta0T[(size_t)(mt*16 + hi*8 + i)*BATCH + b0 + n] = cc[i];
  }
}

// ---------------- ODE RHS: k = W2 @ tanh(W1 @ y + b1) + b2 ----------------
__device__ __forceinline__ void ode_f(const v8f yin[4], v8f kout[4],
    const v16h (*a1)[2], const v16h (*a2)[2],
    const v8f* b1t, const v8f* b2t, int hi)
{
  v16h q0 = make_bfrag(yin[0], yin[1], hi);
  v16h q1 = make_bfrag(yin[2], yin[3], hi);
  v8f h[4];
  #pragma unroll
  for(int m=0;m<4;++m){
    v8f c = b1t[m];
    c = wmma_f16(a1[m][0], q0, c);
    c = wmma_f16(a1[m][1], q1, c);
    #pragma unroll
    for(int i=0;i<8;++i) c[i] = ftanh(c[i]);
    h[m] = c;
  }
  v16h p0 = make_bfrag(h[0], h[1], hi);
  v16h p1 = make_bfrag(h[2], h[3], hi);
  #pragma unroll
  for(int m=0;m<4;++m){
    v8f c = b2t[m];
    c = wmma_f16(a2[m][0], p0, c);
    c = wmma_f16(a2[m][1], p1, c);
    kout[m] = c;
  }
}

// ---------------- ODE (50 x dopri5) + regressor ----------------
__global__ __launch_bounds__(128, 1) void ode_reg_kernel(
    const float* __restrict__ theta0T, const float* __restrict__ tspan,
    const _Float16* __restrict__ o1h, const float* __restrict__ o1b,
    const _Float16* __restrict__ o2h, const float* __restrict__ o2b,
    const _Float16* __restrict__ r1h, const float* __restrict__ r1b,
    const float* __restrict__ r2w, const float* __restrict__ r2b,
    float* __restrict__ out)
{
  const int lane = threadIdx.x & 31;
  const int wv   = threadIdx.x >> 5;
  const int n  = lane & 15;
  const int hi = lane >> 4;
  const int lm = n;
  const int b0 = (blockIdx.x*4 + wv) * 16;

  v16h a1[4][2], a2[4][2];
  v8f  b1t[4], b2t[4];
  #pragma unroll
  for(int m=0;m<4;++m){
    a1[m][0] = load_afrag(o1h, HID, m, 0, lm, hi);
    a1[m][1] = load_afrag(o1h, HID, m, 1, lm, hi);
    a2[m][0] = load_afrag(o2h, HID, m, 0, lm, hi);
    a2[m][1] = load_afrag(o2h, HID, m, 1, lm, hi);
    b1t[m] = bias_tile(o1b, m, hi);
    b2t[m] = bias_tile(o2b, m, hi);
  }

  v8f y[4];
  #pragma unroll
  for(int m=0;m<4;++m)
    #pragma unroll
    for(int i=0;i<8;++i)
      y[m][i] = theta0T[(size_t)(m*16 + hi*8 + i)*BATCH + b0 + n];

  const float dt = (tspan[1] - tspan[0]) * (1.0f/STEPS);

  const float A21 = 0.2f;
  const float A31 = 3.f/40.f,       A32 = 9.f/40.f;
  const float A41 = 44.f/45.f,      A42 = -56.f/15.f,     A43 = 32.f/9.f;
  const float A51 = 19372.f/6561.f, A52 = -25360.f/2187.f, A53 = 64448.f/6561.f, A54 = -212.f/729.f;
  const float A61 = 9017.f/3168.f,  A62 = -355.f/33.f,     A63 = 46732.f/5247.f, A64 = 49.f/176.f, A65 = -5103.f/18656.f;
  const float B1 = 35.f/384.f, B3 = 500.f/1113.f, B4 = 125.f/192.f, B5 = -2187.f/6784.f, B6 = 11.f/84.f;

  v8f k1[4],k2[4],k3[4],k4[4],k5[4],k6[4],yi[4];

  #pragma unroll 1
  for(int st=0; st<STEPS; ++st){
    ode_f(y, k1, a1, a2, b1t, b2t, hi);
    #pragma unroll
    for(int m=0;m<4;++m)
      #pragma unroll
      for(int i=0;i<8;++i) yi[m][i] = y[m][i] + dt*(A21*k1[m][i]);
    ode_f(yi, k2, a1, a2, b1t, b2t, hi);
    #pragma unroll
    for(int m=0;m<4;++m)
      #pragma unroll
      for(int i=0;i<8;++i) yi[m][i] = y[m][i] + dt*(A31*k1[m][i] + A32*k2[m][i]);
    ode_f(yi, k3, a1, a2, b1t, b2t, hi);
    #pragma unroll
    for(int m=0;m<4;++m)
      #pragma unroll
      for(int i=0;i<8;++i) yi[m][i] = y[m][i] + dt*(A41*k1[m][i] + A42*k2[m][i] + A43*k3[m][i]);
    ode_f(yi, k4, a1, a2, b1t, b2t, hi);
    #pragma unroll
    for(int m=0;m<4;++m)
      #pragma unroll
      for(int i=0;i<8;++i) yi[m][i] = y[m][i] + dt*(A51*k1[m][i] + A52*k2[m][i] + A53*k3[m][i] + A54*k4[m][i]);
    ode_f(yi, k5, a1, a2, b1t, b2t, hi);
    #pragma unroll
    for(int m=0;m<4;++m)
      #pragma unroll
      for(int i=0;i<8;++i) yi[m][i] = y[m][i] + dt*(A61*k1[m][i] + A62*k2[m][i] + A63*k3[m][i] + A64*k4[m][i] + A65*k5[m][i]);
    ode_f(yi, k6, a1, a2, b1t, b2t, hi);
    #pragma unroll
    for(int m=0;m<4;++m)
      #pragma unroll
      for(int i=0;i<8;++i)
        y[m][i] = y[m][i] + dt*(B1*k1[m][i] + B3*k3[m][i] + B4*k4[m][i] + B5*k5[m][i] + B6*k6[m][i]);
  }

  // regressor: relu(reg1 @ y + b) then dot with reg2
  v16h qy0 = make_bfrag(y[0], y[1], hi);
  v16h qy1 = make_bfrag(y[2], y[3], hi);
  float partial = 0.0f;
  #pragma unroll
  for(int t=0;t<2;++t){
    v8f c = bias_tile(r1b, t, hi);
    c = wmma_f16(load_afrag(r1h, HID, t, 0, lm, hi), qy0, c);
    c = wmma_f16(load_afrag(r1h, HID, t, 1, lm, hi), qy1, c);
    #pragma unroll
    for(int i=0;i<8;++i){
      float rv = c[i] > 0.0f ? c[i] : 0.0f;
      partial += rv * r2w[t*16 + hi*8 + i];
    }
  }
  float tot = partial + __shfl_xor(partial, 16, 32);
  if (lane < 16) out[b0 + n] = tot + r2b[0];
}

// ---------------- host ----------------
extern "C" void kernel_launch(void* const* d_in, const int* in_sizes, int n_in,
                              void* d_out, int out_size, void* d_ws, size_t ws_size,
                              hipStream_t stream)
{
  (void)in_sizes; (void)n_in; (void)out_size; (void)ws_size;
  const float* x     = (const float*)d_in[0];
  const float* tspan = (const float*)d_in[1];
  const float* convw = (const float*)d_in[2];
  const float* convb = (const float*)d_in[3];
  const float* e1w   = (const float*)d_in[4];
  const float* e1b   = (const float*)d_in[5];
  const float* e2w   = (const float*)d_in[6];
  const float* e2b   = (const float*)d_in[7];
  const float* o1w   = (const float*)d_in[8];
  const float* o1b   = (const float*)d_in[9];
  const float* o2w   = (const float*)d_in[10];
  const float* o2b   = (const float*)d_in[11];
  const float* r1w   = (const float*)d_in[12];
  const float* r1b   = (const float*)d_in[13];
  const float* r2w   = (const float*)d_in[14];
  const float* r2b   = (const float*)d_in[15];
  float* out = (float*)d_out;

  char* ws = (char*)d_ws;
  size_t off = 0;
  float*    theta0T = (float*)(ws + off);    off += (size_t)HID*BATCH*4;   // 16 MB
  _Float16* e1h     = (_Float16*)(ws + off); off += (size_t)ENCD*KFLAT*2;  // 360 KB
  _Float16* wA      = (_Float16*)(ws + off); off += 48*96*2;
  _Float16* e2h     = (_Float16*)(ws + off); off += HID*ENCD*2;
  _Float16* o1h     = (_Float16*)(ws + off); off += HID*HID*2;
  _Float16* o2h     = (_Float16*)(ws + off); off += HID*HID*2;
  _Float16* r1h     = (_Float16*)(ws + off); off += 32*HID*2;

  cvt_f16_kernel<<<(ENCD*KFLAT + 255)/256, 256, 0, stream>>>(e1w, e1h, ENCD*KFLAT);
  cvt_f16_kernel<<<(HID*ENCD + 255)/256, 256, 0, stream>>>(e2w, e2h, HID*ENCD);
  cvt_f16_kernel<<<(HID*HID + 255)/256, 256, 0, stream>>>(o1w, o1h, HID*HID);
  cvt_f16_kernel<<<(HID*HID + 255)/256, 256, 0, stream>>>(o2w, o2h, HID*HID);
  cvt_f16_kernel<<<(32*HID + 255)/256, 256, 0, stream>>>(r1w, r1h, 32*HID);
  prep_convw_kernel<<<(48*96 + 255)/256, 256, 0, stream>>>(convw, wA);

  frontend_kernel<<<BATCH/16, 32, 0, stream>>>(x, convb, wA, e1h, e1b, e2h, e2b, theta0T);

  ode_reg_kernel<<<BATCH/64, 128, 0, stream>>>(theta0T, tspan,
      o1h, o1b, o2h, o2b, r1h, r1b, r2w, r2b, out);
}
